// Transformer_773094113834
// MI455X (gfx1250) — matrix operand order repaired and checked by reading, not compiled, not run on hardware
//
#include <hip/hip_runtime.h>
#include <hip/hip_bf16.h>

// ---------------------------------------------------------------------------
// Types
// ---------------------------------------------------------------------------
typedef _Float16 half_t;
typedef __attribute__((ext_vector_type(16))) _Float16 v16h;
typedef __attribute__((ext_vector_type(8)))  _Float16 v8h;
typedef __attribute__((ext_vector_type(2)))  _Float16 v2h;
typedef __attribute__((ext_vector_type(8)))  float    v8f;
typedef __attribute__((ext_vector_type(4)))  unsigned int v4u;
typedef __attribute__((ext_vector_type(8)))  int      v8i;
typedef __attribute__((ext_vector_type(4)))  int      v4i;

static constexpr int NNODES = 16384;
static constexpr int KNBR   = 16;
static constexpr int IN_D   = 256;
static constexpr int HID    = 512;
static constexpr int OUT_D  = 512;

// LDS B panel: 64 rows x (Kdim + 8) halfs; pad of 16 B per 1024 B row breaks
// bank conflicts on the strided fragment reads (1040/4 mod 64 == 4).
static constexpr int BROWS     = 64;
static constexpr int RS_MAX    = HID + 8;          // 520 halfs (K=512 worst case)

// ---------------------------------------------------------------------------
// Conversions
// ---------------------------------------------------------------------------
__global__ void f32_to_f16_kernel(const float* __restrict__ src,
                                  half_t* __restrict__ dst, int n) {
    int i = blockIdx.x * blockDim.x + threadIdx.x;
    if (i < n) dst[i] = (half_t)src[i];
}

// src: [I, O] row-major f32  ->  dst: [O, I] row-major f16
__global__ void transpose_f32_to_f16_kernel(const float* __restrict__ src,
                                            half_t* __restrict__ dst,
                                            int I, int O) {
    int idx = blockIdx.x * blockDim.x + threadIdx.x;
    if (idx < I * O) {
        int i = idx / O;
        int o = idx - i * O;
        dst[(size_t)o * I + i] = (half_t)src[idx];
    }
}

// ---------------------------------------------------------------------------
// WMMA GEMM:  C[M, Ndim] = epi(A[M, Kdim] (f16) @ Wt[Ndim, Kdim]^T (f16) + bias)
//   block = 256 threads = 8 waves; block tile = 256 rows x 64 cols
//   wave tile = 32 rows x 64 cols -> 8 x v_wmma_f32_16x16x32_f16 per K-step
//   B panel (64 x Kdim f16) staged once into LDS via TENSOR_LOAD_TO_LDS (TDM),
//   padded 16 B per row for conflict-free ds reads.
// EPI: 0 = none, 1 = ELU.  OUTH: 1 = store f16, 0 = store f32.
// ---------------------------------------------------------------------------
template<int EPI, int OUTH>
__global__ __launch_bounds__(256)
void gemm_f16_wmma(const half_t* __restrict__ A,
                   const half_t* __restrict__ Wt,
                   const float*  __restrict__ bias,
                   half_t* __restrict__ Oh,
                   float*  __restrict__ Of,
                   int Kdim, int Ndim) {
    __shared__ half_t Bsh[BROWS * RS_MAX];         // 66,560 B -> 4 blocks/WGP

    const int lane = threadIdx.x & 31;
    const int wave = threadIdx.x >> 5;
    const int lo   = lane & 15;     // N (or M) index within 16-wide fragment
    const int hi   = lane >> 4;     // K sub-block select per ISA 16-bit layouts

    const int m0    = blockIdx.y * 256 + wave * 32;   // wave's 32-row tile base
    const int ncol0 = blockIdx.x * 64;                // block's 64-col tile base
    const int RS    = Kdim + 8;                       // LDS row stride (halfs)

    // ---- stage B panel Wt[ncol0 .. ncol0+64) x [0, Kdim) into LDS ----------
#if __has_builtin(__builtin_amdgcn_tensor_load_to_lds)
    if (threadIdx.x < 32) {
        // D# descriptor, ISA ch.8. Units of data_size = 8 bytes.
        const unsigned units8 = (unsigned)(Kdim / 4);        // row length, 8B units
        const unsigned pad_iv = (Kdim == 512) ? 7u : 6u;     // 256 / 128 DWORDs
        const unsigned long long gaddr =
            (unsigned long long)(const void*)(Wt + (size_t)ncol0 * Kdim);
        const unsigned lds_base = (unsigned)(unsigned long long)(const void*)&Bsh[0];

        v4u g0;
        g0[0] = 1u;                                          // count = 1 (valid)
        g0[1] = lds_base;                                    // lds_addr (bytes)
        g0[2] = (unsigned)(gaddr & 0xffffffffull);           // global_addr[31:0]
        g0[3] = (unsigned)((gaddr >> 32) & 0x1ffffffull)     // global_addr[56:32]
                | (2u << 30);                                // type = 2 (image)

        v8i g1;
        g1[0] = (int)((3u << 16)                             // data_size = 8 B
                      | (1u << 20)                           // pad_enable
                      | (pad_iv << 22)                       // pad_interval
                      | (3u << 25));                         // pad_amount = 4 DW (16 B)
        g1[1] = (int)((units8 & 0xffffu) << 16);             // tensor_dim0[15:0]
        g1[2] = (int)(64u << 16);                            // tensor_dim1 = 64 rows
        g1[3] = (int)((units8 & 0xffffu) << 16);             // tile_dim0
        g1[4] = 64;                                          // tile_dim1 = 64 rows
        g1[5] = (int)units8;                                 // tensor_dim0_stride
        g1[6] = 0;
        g1[7] = 0;

        v4i gz = {};
#if defined(__clang_major__) && (__clang_major__ >= 23)
        v8i z8 = {};
        __builtin_amdgcn_tensor_load_to_lds(g0, g1, gz, gz, z8, 0);
#else
        __builtin_amdgcn_tensor_load_to_lds(g0, g1, gz, gz, 0);
#endif
        __builtin_amdgcn_s_wait_tensorcnt(0);
    }
#else
    {   // Fallback: cooperative copy (same padded layout).
        const int chunks_per_row = Kdim / 8;
        const int total = BROWS * chunks_per_row;
        for (int c = threadIdx.x; c < total; c += 256) {
            const int row = c / chunks_per_row;
            const int col = c - row * chunks_per_row;
            *(v8h*)&Bsh[row * RS + col * 8] =
                *(const v8h*)(Wt + (size_t)(ncol0 + row) * Kdim + col * 8);
        }
    }
#endif
    __syncthreads();

    // ---- main K loop -------------------------------------------------------
    v8f acc[2][4] = {};

    const half_t* Arow0 = A + (size_t)(m0 + lo) * Kdim + hi * 8;        // rows m0..m0+15
    const half_t* Arow1 = A + (size_t)(m0 + 16 + lo) * Kdim + hi * 8;   // rows m0+16..+31

    for (int k0 = 0; k0 < Kdim; k0 += 32) {
        // A fragments (16-bit A 16x32 layout): lane = row, K = hi*8 + [0..8)
        // in v0..v3 and K = hi*8 + 16 + [0..8) in v4..v7.
        union { v16h v; v8h h[2]; } a0, a1;
        a0.h[0] = *(const v8h*)(Arow0 + k0);
        a0.h[1] = *(const v8h*)(Arow0 + k0 + 16);
        a1.h[0] = *(const v8h*)(Arow1 + k0);
        a1.h[1] = *(const v8h*)(Arow1 + k0 + 16);

#pragma unroll
        for (int t = 0; t < 4; ++t) {
            // B fragment from LDS: lane = column n = t*16+lo, K = hi*8 + [0..8) and hi*8 + 16 + [0..8).
            const int bbase = (t * 16 + lo) * RS + k0 + hi * 8;
            union { v16h v; v8h h[2]; } b;
            b.h[0] = *(const v8h*)&Bsh[bbase];
            b.h[1] = *(const v8h*)&Bsh[bbase + 16];

            acc[0][t] = __builtin_amdgcn_wmma_f32_16x16x32_f16(
                false, a0.v, false, b.v, (short)0, acc[0][t], false, false);
            acc[1][t] = __builtin_amdgcn_wmma_f32_16x16x32_f16(
                false, a1.v, false, b.v, (short)0, acc[1][t], false, false);
        }
    }

    // ---- epilogue. C/D layout: lane = column n; rows m = hi*8 + r ----------
#pragma unroll
    for (int mt = 0; mt < 2; ++mt) {
#pragma unroll
        for (int t = 0; t < 4; ++t) {
            const int n = ncol0 + t * 16 + lo;
            const float bn = bias[n];
#pragma unroll
            for (int r = 0; r < 8; ++r) {
                float v = acc[mt][t][r] + bn;
                if (EPI == 1) v = (v > 0.f) ? v : (__expf(v) - 1.f);
                const size_t m = (size_t)(m0 + mt * 16 + hi * 8 + r);
                if (OUTH) Oh[m * Ndim + n] = (half_t)v;
                else      Of[m * Ndim + n] = v;
            }
        }
    }
}

// ---------------------------------------------------------------------------
// LN1: local = elu(layernorm(h2 + 2*pd)) -> f16.  One wave per row of 512.
// ---------------------------------------------------------------------------
__global__ __launch_bounds__(256)
void ln1_kernel(const float* __restrict__ h2, const float* __restrict__ pd,
                const float* __restrict__ g,  const float* __restrict__ b,
                half_t* __restrict__ outh) {
    const int lane = threadIdx.x & 31;
    const int wave = threadIdx.x >> 5;
    const size_t row = (size_t)blockIdx.x * 8 + wave;
    const float* X = h2 + row * HID;
    const float* P = pd + row * HID;

    float x[16], s = 0.f, sq = 0.f;
#pragma unroll
    for (int i = 0; i < 16; ++i) {
        const int j = lane + i * 32;
        const float v = X[j] + 2.f * P[j];
        x[i] = v; s += v; sq += v * v;
    }
#pragma unroll
    for (int off = 16; off > 0; off >>= 1) {
        s  += __shfl_xor(s,  off, 32);
        sq += __shfl_xor(sq, off, 32);
    }
    const float mean = s * (1.f / HID);
    const float var  = sq * (1.f / HID) - mean * mean;
    const float rstd = rsqrtf(var + 1e-5f);

    half_t* O = outh + row * HID;
#pragma unroll
    for (int i = 0; i < 16; ++i) {
        const int j = lane + i * 32;
        float y = (x[i] - mean) * rstd * g[j] + b[j];
        y = (y > 0.f) ? y : (__expf(y) - 1.f);
        O[j] = (half_t)y;
    }
}

// ---------------------------------------------------------------------------
// LN2: out = layernorm(interaction + pipd) -> f32 (final output)
// ---------------------------------------------------------------------------
__global__ __launch_bounds__(256)
void ln2_kernel(const float* __restrict__ ia, const float* __restrict__ pi,
                const float* __restrict__ g,  const float* __restrict__ b,
                float* __restrict__ out) {
    const int lane = threadIdx.x & 31;
    const int wave = threadIdx.x >> 5;
    const size_t row = (size_t)blockIdx.x * 8 + wave;
    const float* X = ia + row * OUT_D;
    const float* P = pi + row * OUT_D;

    float x[16], s = 0.f, sq = 0.f;
#pragma unroll
    for (int i = 0; i < 16; ++i) {
        const int j = lane + i * 32;
        const float v = X[j] + P[j];
        x[i] = v; s += v; sq += v * v;
    }
#pragma unroll
    for (int off = 16; off > 0; off >>= 1) {
        s  += __shfl_xor(s,  off, 32);
        sq += __shfl_xor(sq, off, 32);
    }
    const float mean = s * (1.f / OUT_D);
    const float var  = sq * (1.f / OUT_D) - mean * mean;
    const float rstd = rsqrtf(var + 1e-5f);

    float* O = out + row * OUT_D;
#pragma unroll
    for (int i = 0; i < 16; ++i) {
        const int j = lane + i * 32;
        O[j] = (x[i] - mean) * rstd * g[j] + b[j];
    }
}

// ---------------------------------------------------------------------------
// Neighbor multi-head attention.
//   block = 256 threads = 8 waves = 8 heads; one block per node.
//   lane owns 2 of the 64 head-dims (v2h). Scores via wave32 butterfly.
//   k/v tables are 16 MB f16 each -> gather hits L2 (192 MB).
// ---------------------------------------------------------------------------
__global__ __launch_bounds__(256)
void attn_kernel(const half_t* __restrict__ Q, const half_t* __restrict__ Kt,
                 const half_t* __restrict__ Vt, const int* __restrict__ nbr,
                 half_t* __restrict__ Ob) {
    const int lane = threadIdx.x & 31;
    const int head = threadIdx.x >> 5;
    const int node = blockIdx.x;

    const size_t hb = (size_t)node * HID + head * 64;
    const v2h qv = ((const v2h*)(Q + hb))[lane];
    const float q0 = (float)qv[0], q1 = (float)qv[1];

    int idx[KNBR];
    const int* nb = nbr + (size_t)node * KNBR;
#pragma unroll
    for (int k = 0; k < KNBR; ++k) idx[k] = nb[k];

    float sc[KNBR];
#pragma unroll
    for (int k = 0; k < KNBR; ++k) {
        const v2h kv = ((const v2h*)(Kt + (size_t)idx[k] * HID + head * 64))[lane];
        float p = q0 * (float)kv[0] + q1 * (float)kv[1];
#pragma unroll
        for (int off = 16; off > 0; off >>= 1) p += __shfl_xor(p, off, 32);
        sc[k] = p * 0.125f;                    // 1/sqrt(64)
    }

    float mx = sc[0];
#pragma unroll
    for (int k = 1; k < KNBR; ++k) mx = fmaxf(mx, sc[k]);
    float ssum = 0.f;
#pragma unroll
    for (int k = 0; k < KNBR; ++k) { sc[k] = __expf(sc[k] - mx); ssum += sc[k]; }
    const float inv = 1.f / ssum;

    float a0 = 0.f, a1 = 0.f;
#pragma unroll
    for (int k = 0; k < KNBR; ++k) {
        const v2h vv = ((const v2h*)(Vt + (size_t)idx[k] * HID + head * 64))[lane];
        const float w = sc[k] * inv;
        a0 += w * (float)vv[0];
        a1 += w * (float)vv[1];
    }
    v2h o; o[0] = (half_t)a0; o[1] = (half_t)a1;
    ((v2h*)(Ob + hb))[lane] = o;
}

// ---------------------------------------------------------------------------
// Host-side orchestration
// ---------------------------------------------------------------------------
extern "C" void kernel_launch(void* const* d_in, const int* in_sizes, int n_in,
                              void* d_out, int out_size, void* d_ws, size_t ws_size,
                              hipStream_t stream) {
    const float* data      = (const float*)d_in[0];
    const int*   neighbors = (const int*)  d_in[1];
    const float* mlp_w0 = (const float*)d_in[2];  const float* mlp_b0 = (const float*)d_in[3];
    const float* mlp_w1 = (const float*)d_in[4];  const float* mlp_b1 = (const float*)d_in[5];
    const float* mlp_w2 = (const float*)d_in[6];  const float* mlp_b2 = (const float*)d_in[7];
    const float* pd_w   = (const float*)d_in[8];  const float* pd_b   = (const float*)d_in[9];
    const float* pi_w   = (const float*)d_in[10]; const float* pi_b   = (const float*)d_in[11];
    const float* q_w    = (const float*)d_in[12]; const float* q_b    = (const float*)d_in[13];
    const float* k_w    = (const float*)d_in[14]; const float* k_b    = (const float*)d_in[15];
    const float* v_w    = (const float*)d_in[16]; const float* v_b    = (const float*)d_in[17];
    const float* o_w    = (const float*)d_in[18]; const float* o_b    = (const float*)d_in[19];
    const float* ln1_g  = (const float*)d_in[20]; const float* ln1_b  = (const float*)d_in[21];
    const float* ln2_g  = (const float*)d_in[22]; const float* ln2_b  = (const float*)d_in[23];

    char* ws = (char*)d_ws;
    size_t off = 0;
    auto carve = [&](size_t bytes) -> char* {
        char* p = ws + off;
        off += (bytes + 255) & ~(size_t)255;
        return p;
    };

    half_t* data_h = (half_t*)carve((size_t)NNODES * IN_D * 2);
    half_t* w0t = (half_t*)carve((size_t)IN_D * HID * 2);
    half_t* w1t = (half_t*)carve((size_t)HID  * HID * 2);
    half_t* w2t = (half_t*)carve((size_t)HID  * HID * 2);
    half_t* pdt = (half_t*)carve((size_t)IN_D * HID * 2);
    half_t* pit = (half_t*)carve((size_t)IN_D * OUT_D * 2);
    half_t* qt  = (half_t*)carve((size_t)HID  * HID * 2);
    half_t* kt  = (half_t*)carve((size_t)HID  * HID * 2);
    half_t* vt  = (half_t*)carve((size_t)HID  * HID * 2);
    half_t* ot  = (half_t*)carve((size_t)HID  * OUT_D * 2);
    half_t* B0  = (half_t*)carve((size_t)NNODES * HID * 2);
    half_t* B1  = (half_t*)carve((size_t)NNODES * HID * 2);
    half_t* B2  = (half_t*)carve((size_t)NNODES * HID * 2);
    half_t* B3  = (half_t*)carve((size_t)NNODES * HID * 2);
    float*  F0  = (float*) carve((size_t)NNODES * HID * 4);
    float*  F1  = (float*) carve((size_t)NNODES * HID * 4);
    (void)ws_size; (void)n_in; (void)in_sizes; (void)out_size;

    // --- conversions ---
    {
        int n = NNODES * IN_D;
        f32_to_f16_kernel<<<(n + 255) / 256, 256, 0, stream>>>(data, data_h, n);
    }
    auto xp = [&](const float* src, half_t* dst, int I, int O) {
        int n = I * O;
        transpose_f32_to_f16_kernel<<<(n + 255) / 256, 256, 0, stream>>>(src, dst, I, O);
    };
    xp(mlp_w0, w0t, IN_D, HID);
    xp(mlp_w1, w1t, HID,  HID);
    xp(mlp_w2, w2t, HID,  HID);
    xp(pd_w,   pdt, IN_D, HID);
    xp(pi_w,   pit, IN_D, OUT_D);
    xp(q_w,    qt,  HID,  HID);
    xp(k_w,    kt,  HID,  HID);
    xp(v_w,    vt,  HID,  HID);
    xp(o_w,    ot,  HID,  OUT_D);

    const dim3 gemm_grid(HID / 64, NNODES / 256);   // (8, 64)
    const dim3 gemm_block(256);

    // --- local MLP branch ---
    gemm_f16_wmma<1, 1><<<gemm_grid, gemm_block, 0, stream>>>(
        data_h, w0t, mlp_b0, B0, nullptr, IN_D, HID);            // h0 = elu(...)
    gemm_f16_wmma<1, 1><<<gemm_grid, gemm_block, 0, stream>>>(
        B0, w1t, mlp_b1, B1, nullptr, HID, HID);                 // h1 = elu(...)
    gemm_f16_wmma<0, 0><<<gemm_grid, gemm_block, 0, stream>>>(
        B1, w2t, mlp_b2, nullptr, F0, HID, HID);                 // h2 (f32)
    gemm_f16_wmma<0, 0><<<gemm_grid, gemm_block, 0, stream>>>(
        data_h, pdt, pd_b, nullptr, F1, IN_D, HID);              // pd (f32)

    // local = elu(LN(h2 + 2*pd))  (since local+pd = h+2pd)
    ln1_kernel<<<NNODES / 8, 256, 0, stream>>>(F0, F1, ln1_g, ln1_b, B2);

    // --- q, k, v projections (f16 outputs) ---
    gemm_f16_wmma<0, 1><<<gemm_grid, gemm_block, 0, stream>>>(
        B2, qt, q_b, B0, nullptr, HID, HID);
    gemm_f16_wmma<0, 1><<<gemm_grid, gemm_block, 0, stream>>>(
        B2, kt, k_b, B1, nullptr, HID, HID);
    gemm_f16_wmma<0, 1><<<gemm_grid, gemm_block, 0, stream>>>(
        B2, vt, v_b, B3, nullptr, HID, HID);

    // --- neighbor attention (overwrites B2 = attn, local no longer needed) ---
    attn_kernel<<<NNODES, 256, 0, stream>>>(B0, B1, B3, neighbors, B2);

    // --- output projection + residual + LN ---
    gemm_f16_wmma<0, 0><<<gemm_grid, gemm_block, 0, stream>>>(
        B2, ot, o_b, nullptr, F0, HID, OUT_D);                   // interaction
    gemm_f16_wmma<0, 0><<<gemm_grid, gemm_block, 0, stream>>>(
        data_h, pit, pi_b, nullptr, F1, IN_D, OUT_D);            // data @ pi_w + pi_b
    ln2_kernel<<<NNODES / 8, 256, 0, stream>>>(F0, F1, ln2_g, ln2_b, (float*)d_out);
}
